// MultiHeadAttention_52158082843071
// MI455X (gfx1250) — matrix operand order repaired and checked by reading, not compiled, not run on hardware
//
#include <hip/hip_runtime.h>
#include <hip/hip_bf16.h>

typedef _Float16 f16;
typedef __attribute__((ext_vector_type(16))) _Float16 v16h;
typedef __attribute__((ext_vector_type(8)))  _Float16 v8h;
typedef __attribute__((ext_vector_type(2)))  _Float16 v2h;
typedef __attribute__((ext_vector_type(8)))  float    v8f;
typedef __attribute__((ext_vector_type(4)))  int      v4i;

#define B_  2
#define T_  2048
#define D_  1024
#define H_  16
#define DH_ 64

__device__ __forceinline__ v16h cat8(v8h lo, v8h hi) {
  v16h r;
#pragma unroll
  for (int i = 0; i < 8; ++i) { r[i] = lo[i]; r[i + 8] = hi[i]; }
  return r;
}

// A-matrix fragment (16x32 f16): lane<16 -> row=lane, K [0..15];
// lane>=16 -> row=lane-16, K [16..31].  rowp = base of this
// lane's row; c0 = (lane<16 ? 0 : 8).
__device__ __forceinline__ v16h frag_a(const f16* rowp, int c0) {
  v8h lo = *(const v8h*)(rowp + 2 * c0);
  v8h hi = *(const v8h*)(rowp + 2 * c0 + 8);
  return cat8(lo, hi);
}

// B-matrix fragment (32x16 f16): lane holds col = lane%16, 16 contiguous K
// values starting at (lane/16)*16.  p = address of this lane's first K value.
__device__ __forceinline__ v16h frag_b(const f16* p) {
  v8h lo = *(const v8h*)p;
  v8h hi = *(const v8h*)(p + 8);
  return cat8(lo, hi);
}

__device__ __forceinline__ v8f wmma_f16(v16h a, v16h b, v8f c) {
  return __builtin_amdgcn_wmma_f32_16x16x32_f16(false, a, false, b, (short)0, c,
                                                false, false);
}

__device__ __forceinline__ v8f zero8() {
  v8f z = {0.f, 0.f, 0.f, 0.f, 0.f, 0.f, 0.f, 0.f};
  return z;
}

// ---- CDNA5 async global->LDS copy (16B per lane), with portable fallback ----
#if defined(__gfx1250__) && __has_builtin(__builtin_amdgcn_global_load_async_to_lds_b128)
#define HAVE_ASYNC_LDS 1
typedef __attribute__((address_space(1))) v4i* as1_v4i_p;
typedef __attribute__((address_space(3))) v4i* as3_v4i_p;
#endif

__device__ __forceinline__ void cp16_async(const f16* g, f16* l) {
#ifdef HAVE_ASYNC_LDS
  __builtin_amdgcn_global_load_async_to_lds_b128((as1_v4i_p)g, (as3_v4i_p)l, 0,
                                                 0);
#else
  *(v8h*)l = *(const v8h*)g;
#endif
}

__device__ __forceinline__ void cp_async_wait() {
#ifdef HAVE_ASYNC_LDS
#if __has_builtin(__builtin_amdgcn_s_wait_asynccnt)
  __builtin_amdgcn_s_wait_asynccnt(0);
#else
  asm volatile("s_wait_asynccnt 0" ::: "memory");
#endif
#endif
}

// ------------------------- f32 -> f16 conversion -------------------------
__global__ void mha_cvt_f16(const float* __restrict__ in, f16* __restrict__ out,
                            int n4) {
  int i = blockIdx.x * blockDim.x + threadIdx.x;
  if (i < n4) {
    float4 v = ((const float4*)in)[i];
    f16* o = out + (size_t)i * 4;
    o[0] = (f16)v.x; o[1] = (f16)v.y; o[2] = (f16)v.z; o[3] = (f16)v.w;
  }
}

// ------------------------- tiled WMMA GEMM core -------------------------
// Y[M,N] = A[M,K](f16) @ W[N,K]^T(f16) + bias(f32)
// block: 256 threads = 8 waves (2 x 4), block tile 64(M) x 128(N), K step 32,
// double-buffered async LDS staging of the A tile.
// MODE 0: scatter into q[B,H,T,DH], k[B,H,T,DH], vt[B,H,DH,T]  (f16)
// MODE 1: write f32 to out0[M,N]
template <int MODE>
__global__ __launch_bounds__(256) void mha_gemm(
    const f16* __restrict__ A, const f16* __restrict__ W,
    const float* __restrict__ bias, float* __restrict__ out0,
    f16* __restrict__ q, f16* __restrict__ k, f16* __restrict__ vt, int M,
    int N, int K) {
  __shared__ __attribute__((aligned(16))) f16 As[2][64 * 32];

  const int tid  = threadIdx.x;
  const int lane = tid & 31;
  const int w    = tid >> 5;
  const int m0   = blockIdx.y * 64;
  const int n0   = blockIdx.x * 128;
  const int mw   = (w >> 2) * 32;   // wave row offset in block tile
  const int nw   = (w & 3) * 32;    // wave col offset in block tile
  const int rsel = lane & 15;
  const int hsel = lane >> 4;
  const int c0   = hsel * 8;

  v8f acc[2][2];
#pragma unroll
  for (int i = 0; i < 2; ++i)
#pragma unroll
    for (int j = 0; j < 2; ++j) acc[i][j] = zero8();

  const int srow = tid >> 2;         // 0..63
  const int sseg = (tid & 3) * 8;    // 0,8,16,24 (halfs)
  const f16* aNext = A + (size_t)(m0 + srow) * K + sseg;

  // prologue: stage first A tile
  cp16_async(aNext, &As[0][srow * 32 + sseg]);

  int cur = 0;
  for (int kk = 0; kk < K; kk += 32, cur ^= 1) {
    cp_async_wait();
    __syncthreads();
    if (kk + 32 < K)  // prefetch next tile into the other buffer
      cp16_async(aNext + kk + 32, &As[cur ^ 1][srow * 32 + sseg]);

    v16h af[2], bf[2];
#pragma unroll
    for (int ms = 0; ms < 2; ++ms)
      af[ms] = frag_a(&As[cur][(mw + ms * 16 + rsel) * 32], c0);
#pragma unroll
    for (int ns = 0; ns < 2; ++ns) {
      const int n      = n0 + nw + ns * 16 + rsel;
      const f16* bp    = W + (size_t)n * K + kk + hsel * 16;
      bf[ns]           = frag_b(bp);
      __builtin_prefetch(bp + 32, 0, 1);  // next K-step -> global_prefetch_b8
    }
#pragma unroll
    for (int ms = 0; ms < 2; ++ms)
#pragma unroll
      for (int ns = 0; ns < 2; ++ns)
        acc[ms][ns] = wmma_f16(af[ms], bf[ns], acc[ms][ns]);
  }

  // epilogue: C layout => lane holds col n (fixed), rows r + 8*hsel
  const int hoff = hsel * 8;
#pragma unroll
  for (int ms = 0; ms < 2; ++ms) {
#pragma unroll
    for (int ns = 0; ns < 2; ++ns) {
      const int n  = n0 + nw + ns * 16 + rsel;
      const float bv = bias[n];
      if (MODE == 1) {
#pragma unroll
        for (int r = 0; r < 8; ++r) {
          const int m = m0 + mw + ms * 16 + r + hoff;
          out0[(size_t)m * N + n] = acc[ms][ns][r] + bv;
        }
      } else {
        const int which = n >> 10;          // 0=q 1=k 2=v
        const int rem   = n & 1023;
        const int head  = rem >> 6;
        const int d     = rem & 63;
#pragma unroll
        for (int r = 0; r < 8; ++r) {
          const int m    = m0 + mw + ms * 16 + r + hoff;
          const int bidx = m >> 11;          // batch
          const int trow = m & 2047;         // token
          const f16 hv   = (f16)(acc[ms][ns][r] + bv);
          const size_t bh = (size_t)bidx * H_ + head;
          if (which == 0)
            q[(bh * T_ + trow) * DH_ + d] = hv;
          else if (which == 1)
            k[(bh * T_ + trow) * DH_ + d] = hv;
          else
            vt[(bh * DH_ + d) * T_ + trow] = hv;   // V stored transposed
        }
      }
    }
  }
}

// ------------------------- flash attention -------------------------
// 1 wave per 16-query tile; 32 keys / iteration; online softmax (f32 state).
// Q pre-scaled by dh^-0.5; row-sums via WMMA against ones; row-max via packed
// f16 butterfly (2 rows per 32-bit reg -> v_pk_max + 16 bpermutes/block);
// only the single diagonal block runs the masked path.
template <bool MASKED>
__device__ __forceinline__ void attn_block(
    int kbase, int qt16, int rsel, int hsel, int c0, int hoff,
    const f16* __restrict__ kb, const f16* __restrict__ vb, f16* pw,
    const v16h& aq0, const v16h& aq1, const v16h& ones, float mi[8], v8f& lacc,
    v8f acc[4]) {
  // ---- S = Q K^T (16 x 32), two 16x16 column tiles ----
  v8f s0 = zero8(), s1 = zero8();
  const f16* krow0 = kb + (size_t)(kbase + rsel) * DH_ + hsel * 16;
  const f16* krow1 = kb + (size_t)(kbase + 16 + rsel) * DH_ + hsel * 16;
  s0 = wmma_f16(aq0, frag_b(krow0), s0);
  s0 = wmma_f16(aq1, frag_b(krow0 + 32), s0);
  s1 = wmma_f16(aq0, frag_b(krow1), s1);
  s1 = wmma_f16(aq1, frag_b(krow1 + 32), s1);

  if (MASKED) {
#pragma unroll
    for (int r = 0; r < 8; ++r) {
      const int row  = qt16 + r + hoff;
      const int col0 = kbase + rsel;
      s0[r] = (col0 <= row)      ? s0[r] : -3.0e38f;
      s1[r] = (col0 + 16 <= row) ? s1[r] : -3.0e38f;
    }
  }

  // ---- block row-max: packed f16, 2 rows per register, 16-lane butterfly.
  // The softmax max is only a stabilizer, so f16 precision is sufficient.
  v2h hp[4];
#pragma unroll
  for (int j = 0; j < 4; ++j) {
    hp[j][0] = (f16)fmaxf(s0[2 * j],     s1[2 * j]);
    hp[j][1] = (f16)fmaxf(s0[2 * j + 1], s1[2 * j + 1]);
  }
#pragma unroll
  for (int m = 1; m < 16; m <<= 1) {
#pragma unroll
    for (int j = 0; j < 4; ++j) {
      const int o = __shfl_xor(__builtin_bit_cast(int, hp[j]), m, 16);
      hp[j] = __builtin_elementwise_max(hp[j], __builtin_bit_cast(v2h, o));
    }
  }

  // ---- online softmax update ----
  float p0[8], p1[8];
#pragma unroll
  for (int r = 0; r < 8; ++r) {
    const float nm   = fmaxf(mi[r], (float)hp[r >> 1][r & 1]);
    const float resc = __expf(mi[r] - nm);
    mi[r] = nm;
    p0[r] = __expf(s0[r] - nm);
    p1[r] = __expf(s1[r] - nm);
    lacc[r] *= resc;
#pragma unroll
    for (int t = 0; t < 4; ++t) acc[t][r] *= resc;
  }

  // ---- P: C-layout -> LDS -> A-layout fragment ----
#pragma unroll
  for (int r = 0; r < 8; ++r) {
    const int Ml = r + hoff;
    pw[Ml * 32 + rsel]      = (f16)p0[r];
    pw[Ml * 32 + 16 + rsel] = (f16)p1[r];
  }
  asm volatile("" ::: "memory");  // DS stores precede DS loads (in-order HW)
  const v16h ap = frag_a(&pw[rsel * 32], c0);

  // ---- row sums: lacc += P @ ones  (replicated across lanes) ----
  lacc = wmma_f16(ap, ones, lacc);

  // ---- ctx += P (16x32) @ V (32x64) ----
#pragma unroll
  for (int t = 0; t < 4; ++t) {
    const f16* vrow = vb + (size_t)(t * 16 + rsel) * T_ + kbase + hsel * 16;
    acc[t] = wmma_f16(ap, frag_b(vrow), acc[t]);
  }
}

#define AW 4
__global__ __launch_bounds__(32 * AW) void mha_attn(
    const f16* __restrict__ q, const f16* __restrict__ k,
    const f16* __restrict__ vt, f16* __restrict__ ctx) {
  __shared__ __attribute__((aligned(16))) f16 pb[AW][16 * 32];

  const int lane  = threadIdx.x & 31;
  const int widx  = threadIdx.x >> 5;
  const int wg    = blockIdx.x * AW + widx;
  const int bh    = wg >> 7;          // T_/16 = 128 q-tiles per (b,h)
  const int qtile = wg & 127;
  const int qt16  = qtile * 16;

  const f16* qb = q  + (size_t)bh * T_ * DH_;
  const f16* kb = k  + (size_t)bh * T_ * DH_;
  const f16* vb = vt + (size_t)bh * DH_ * T_;

  const int rsel = lane & 15;
  const int hsel = lane >> 4;
  const int c0   = hsel * 8;
  const int hoff = hsel * 8;

  // Q fragments (16 x 64 as two 16x32 A-fragments), pre-scaled by dh^-0.5
  v16h sc;
#pragma unroll
  for (int i = 0; i < 16; ++i) sc[i] = (f16)0.125f;
  const f16* qrow = qb + (size_t)(qt16 + rsel) * DH_;
  const v16h aq0 = frag_a(qrow, c0) * sc;        // dh 0..31
  const v16h aq1 = frag_a(qrow + 32, c0) * sc;   // dh 32..63

  v16h ones;
#pragma unroll
  for (int i = 0; i < 16; ++i) ones[i] = (f16)1.0f;

  v8f acc[4];
#pragma unroll
  for (int t = 0; t < 4; ++t) acc[t] = zero8();
  v8f lacc = zero8();
  float mi[8];
#pragma unroll
  for (int r = 0; r < 8; ++r) mi[r] = -3.0e38f;

  f16* pw = pb[widx];

  // causal: keys [0, qt16+16).  Exactly one block straddles the diagonal.
  const int nfull = qt16 >> 5;  // fully-unmasked 32-key blocks
  for (int kbase = 0; kbase < (nfull << 5); kbase += 32)
    attn_block<false>(kbase, qt16, rsel, hsel, c0, hoff, kb, vb, pw, aq0, aq1,
                      ones, mi, lacc, acc);
  attn_block<true>(nfull << 5, qt16, rsel, hsel, c0, hoff, kb, vb, pw, aq0,
                   aq1, ones, mi, lacc, acc);

  // ---- normalize and write ctx (f16, [B*T, D], d = h*64+dh) ----
  const int b = bh >> 4, h = bh & 15;
#pragma unroll
  for (int r = 0; r < 8; ++r) {
    const float inv = 1.0f / lacc[r];
    const int row   = b * T_ + qt16 + r + hoff;
#pragma unroll
    for (int t = 0; t < 4; ++t) {
      const int col = h * DH_ + t * 16 + rsel;
      ctx[(size_t)row * D_ + col] = (f16)(acc[t][r] * inv);
    }
  }
}

// ------------------------- host-side launcher -------------------------
extern "C" void kernel_launch(void* const* d_in, const int* in_sizes, int n_in,
                              void* d_out, int out_size, void* d_ws,
                              size_t ws_size, hipStream_t stream) {
  const float* x      = (const float*)d_in[0];   // [B,T,D]
  const float* qkv_w  = (const float*)d_in[1];   // [3D,D]
  const float* qkv_b  = (const float*)d_in[2];   // [3D]
  const float* out_w  = (const float*)d_in[3];   // [D,D]
  const float* out_b  = (const float*)d_in[4];   // [D]
  float* out          = (float*)d_out;           // [B,T,D]

  const int M = B_ * T_;          // 4096
  const int K = D_;               // 1024
  const int N3 = 3 * D_;          // 3072

  char* ws = (char*)d_ws;
  f16* xh   = (f16*)ws;                 ws += (size_t)M * D_ * 2;      // 8 MB
  f16* wq   = (f16*)ws;                 ws += (size_t)N3 * D_ * 2;     // 6 MB
  f16* wo   = (f16*)ws;                 ws += (size_t)D_ * D_ * 2;     // 2 MB
  f16* qh   = (f16*)ws;                 ws += (size_t)M * D_ * 2;      // 8 MB
  f16* kh   = (f16*)ws;                 ws += (size_t)M * D_ * 2;      // 8 MB
  f16* vth  = (f16*)ws;                 ws += (size_t)M * D_ * 2;      // 8 MB
  f16* ctxh = (f16*)ws;                 ws += (size_t)M * D_ * 2;      // 8 MB

  // 1) convert inputs to f16
  {
    int n4 = M * D_ / 4;
    mha_cvt_f16<<<(n4 + 255) / 256, 256, 0, stream>>>(x, xh, n4);
    n4 = N3 * D_ / 4;
    mha_cvt_f16<<<(n4 + 255) / 256, 256, 0, stream>>>(qkv_w, wq, n4);
    n4 = D_ * D_ / 4;
    mha_cvt_f16<<<(n4 + 255) / 256, 256, 0, stream>>>(out_w, wo, n4);
  }
  // 2) QKV projection -> q/k/vt (f16)
  {
    dim3 grid(N3 / 128, M / 64);
    mha_gemm<0><<<grid, 256, 0, stream>>>(xh, wq, qkv_b, nullptr, qh, kh, vth,
                                          M, N3, K);
  }
  // 3) causal flash attention -> ctx (f16)
  {
    const int waves = B_ * H_ * (T_ / 16);   // 4096
    mha_attn<<<waves / AW, 32 * AW, 0, stream>>>(qh, kh, vth, ctxh);
  }
  // 4) output projection -> f32 d_out
  {
    dim3 grid(D_ / 128, M / 64);
    mha_gemm<1><<<grid, 256, 0, stream>>>(ctxh, wo, out_b, out, nullptr,
                                          nullptr, nullptr, M, D_, K);
  }
  (void)in_sizes; (void)n_in; (void)out_size; (void)ws_size;
}
